// GumbelSoftmaxQuantizer_64166811402863
// MI455X (gfx1250) — matrix operand order repaired and checked by reading, not compiled, not run on hardware
//
#include <hip/hip_runtime.h>
#include <hip/hip_bf16.h>
#include <stdint.h>

typedef __attribute__((ext_vector_type(16))) __bf16 v16bf;
typedef __attribute__((ext_vector_type(8)))  float  v8f;

#define NROWS    32768          // B*S
#define HID      512
#define GRP      2
#define CCNT     320
#define DCODE    128            // VQ_DIM / G
#define GC       640            // G*C
#define TILES    20             // 320 / 16
#define WAVES    4
#define ROWS_PER_WG (WAVES * 16)
#define WS_WQ_OFF 4096          // byte offset of bf16 Wq copy inside d_ws

// pack two floats to packed bf16 dword: 1 perm + 2 adds (round-to-nearest, ties-away)
__device__ __forceinline__ uint32_t packbf(float lo, float hi) {
  uint32_t a = __float_as_uint(lo) + 0x8000u;
  uint32_t b = __float_as_uint(hi) + 0x8000u;
  return __builtin_amdgcn_perm(b, a, 0x07060302u);  // {b[31:16], a[31:16]}
}

__device__ __forceinline__ v16bf cvt16(float4 a, float4 b, float4 c, float4 d) {
  union { v16bf v; uint32_t u[8]; } r;
  r.u[0] = packbf(a.x, a.y); r.u[1] = packbf(a.z, a.w);
  r.u[2] = packbf(b.x, b.y); r.u[3] = packbf(b.z, b.w);
  r.u[4] = packbf(c.x, c.y); r.u[5] = packbf(c.z, c.w);
  r.u[6] = packbf(d.x, d.y); r.u[7] = packbf(d.z, d.w);
  return r.v;
}

// ---------------------------------------------------------------------------
// Pre-convert Wq (fp32 [640,512]) to bf16 (RNE) in workspace; one pair/thread.
// ---------------------------------------------------------------------------
__global__ void gsq_prep(const float* __restrict__ Wq, uint32_t* __restrict__ wqb) {
  int p = blockIdx.x * blockDim.x + threadIdx.x;      // pair index, 163840 total
  float lo = Wq[2 * p], hi = Wq[2 * p + 1];
  uint32_t a = __float_as_uint(lo); a += 0x7FFFu + ((a >> 16) & 1u);
  uint32_t b = __float_as_uint(hi); b += 0x7FFFu + ((b >> 16) & 1u);
  wqb[p] = __builtin_amdgcn_perm(b, a, 0x07060302u);
}

// ---------------------------------------------------------------------------
// Fused: GEMM(logits) -> argmax(l+gumbel) -> softmax stats -> gather.
// Logits live entirely in WMMA accumulator registers (20 x v8f per wave).
// ---------------------------------------------------------------------------
template <bool PRE>
__global__ __launch_bounds__(128, 1)
void gsq_main(const float* __restrict__ x,        // [32768, 512]
              const float* __restrict__ gumbel,   // [65536, 320]
              const float* __restrict__ Wq,       // [640, 512] fp32 (fallback)
              const __bf16* __restrict__ wqb,     // [640, 512] bf16 (PRE path)
              const float* __restrict__ bq,       // [640]
              const float* __restrict__ cv,       // [640, 128]
              float* __restrict__ out,            // [32768, 256] (+ scalar)
              float* __restrict__ meanSum)        // [640]
{
  __shared__ float meanAcc[GC];
  __shared__ int   idxBuf[WAVES][16];

  const int tid  = threadIdx.x;
  const int wave = tid >> 5;
  const int lane = tid & 31;
  const int l16  = lane & 15;
  const bool hi  = (lane >= 16);
  const int rowBase = blockIdx.x * ROWS_PER_WG + wave * 16;

  for (int i = tid; i < GC; i += 128) meanAcc[i] = 0.0f;
  __syncthreads();

  // A fragment: lane holds row (lane&15); K runs [kb,kb+16)
  const int arow   = rowBase + l16;
  const int kbaseA = hi ? 16 : 0;
  const int kbaseB = hi ? 16 : 0;   // B fragment: lane holds col (lane&15), 16 consecutive K

  for (int g = 0; g < GRP; ++g) {
    const int colBase = g * CCNT;

    v8f acc[TILES];
    #pragma unroll
    for (int t = 0; t < TILES; ++t) acc[t] = (v8f){0,0,0,0,0,0,0,0};

    // ---------------- GEMM: 16 rows x 320 cols, K = 512 ----------------
    for (int k0 = 0; k0 < HID; k0 += 32) {
      const float* ap = x + (size_t)arow * HID + k0 + kbaseA;
      v16bf af = cvt16(*(const float4*)(ap),      *(const float4*)(ap + 4),
                       *(const float4*)(ap + 8),  *(const float4*)(ap + 12));
      #pragma unroll
      for (int t = 0; t < TILES; ++t) {
        v16bf bf;
        if constexpr (PRE) {
          bf = *(const v16bf*)(wqb + (size_t)(colBase + t * 16 + l16) * HID
                                   + k0 + kbaseB);
        } else {
          const float* bp = Wq + (size_t)(colBase + t * 16 + l16) * HID + k0 + kbaseB;
          bf = cvt16(*(const float4*)(bp),     *(const float4*)(bp + 4),
                     *(const float4*)(bp + 8), *(const float4*)(bp + 12));
        }
        acc[t] = __builtin_amdgcn_wmma_f32_16x16x32_bf16(
                     false, af, false, bf, (short)0, acc[t], false, false);
      }
    }

    // bias add (column-indexed)
    #pragma unroll
    for (int t = 0; t < TILES; ++t) {
      float bv = bq[colBase + t * 16 + l16];
      #pragma unroll
      for (int v = 0; v < 8; ++v) acc[t][v] += bv;
    }

    // row-wise max (C layout: VGPR v <-> row v / v+8; lane&15 <-> column)
    float m[8];
    #pragma unroll
    for (int v = 0; v < 8; ++v) {
      float mv = acc[0][v];
      #pragma unroll
      for (int t = 1; t < TILES; ++t) mv = fmaxf(mv, acc[t][v]);
      #pragma unroll
      for (int off = 1; off <= 8; off <<= 1)
        mv = fmaxf(mv, __shfl_xor(mv, off, 32));
      m[v] = mv;
    }

    // argmax over (logit + gumbel) — uses raw logits, so do it before exp
    #pragma unroll
    for (int v = 0; v < 8; ++v) {
      int rowv = rowBase + (hi ? v + 8 : v);
      const float* gp = gumbel + ((size_t)rowv * GRP + g) * CCNT + l16;
      float best = -3.4e38f; int bidx = 0;
      #pragma unroll
      for (int t = 0; t < TILES; ++t) {
        float val = acc[t][v] + gp[t * 16];
        int   ci  = t * 16 + l16;
        if (val > best || (val == best && ci < bidx)) { best = val; bidx = ci; }
      }
      #pragma unroll
      for (int off = 1; off <= 8; off <<= 1) {
        float ob = __shfl_xor(best, off, 32);
        int   oi = __shfl_xor(bidx, off, 32);
        if (ob > best || (ob == best && oi < bidx)) { best = ob; bidx = oi; }
      }
      if (l16 == 0) idxBuf[wave][hi ? v + 8 : v] = bidx;
    }

    // exp in place (single transcendental pass), then row sum + prob accumulate
    #pragma unroll
    for (int v = 0; v < 8; ++v) {
      float sv = 0.0f;
      #pragma unroll
      for (int t = 0; t < TILES; ++t) {
        float e = __expf(acc[t][v] - m[v]);
        acc[t][v] = e;
        sv += e;
      }
      #pragma unroll
      for (int off = 1; off <= 8; off <<= 1)
        sv += __shfl_xor(sv, off, 32);
      float rs = 1.0f / sv;
      #pragma unroll
      for (int t = 0; t < TILES; ++t)
        atomicAdd(&meanAcc[colBase + t * 16 + l16], acc[t][v] * rs);
    }
    __syncthreads();

    // gather codevectors: one float4 per lane = 512B coalesced per row
    for (int r = 0; r < 16; ++r) {
      int ii = idxBuf[wave][r];
      const float4* src = (const float4*)(cv + (size_t)(colBase + ii) * DCODE) + lane;
      float4*       dst = (float4*)(out + (size_t)(rowBase + r) * 256 + g * DCODE) + lane;
      *dst = *src;
    }
    __syncthreads();
  }

  // flush block-local softmax sums to global
  for (int i = tid; i < GC; i += 128)
    atomicAdd(&meanSum[i], meanAcc[i]);
}

// ---------------------------------------------------------------------------
__global__ void gsq_zero(float* meanSum) {
  meanSum[threadIdx.x] = 0.0f;
}

__global__ void gsq_finalize(const float* __restrict__ meanSum,
                             float* __restrict__ out, int outOff) {
  __shared__ float H[2];
  int t = threadIdx.x;                // 640 threads: g = t/320, c = t%320
  if (t < 2) H[t] = 0.0f;
  __syncthreads();
  float mp = meanSum[t] * (1.0f / (float)NROWS);
  atomicAdd(&H[t / CCNT], mp * __logf(mp + 1e-7f));
  __syncthreads();
  if (t == 0) out[outOff] = __expf(-H[0]) + __expf(-H[1]);
}

// ---------------------------------------------------------------------------
extern "C" void kernel_launch(void* const* d_in, const int* in_sizes, int n_in,
                              void* d_out, int out_size, void* d_ws, size_t ws_size,
                              hipStream_t stream) {
  const float* x      = (const float*)d_in[0];
  const float* gumbel = (const float*)d_in[1];
  const float* Wq     = (const float*)d_in[2];
  const float* bq     = (const float*)d_in[3];
  const float* cv     = (const float*)d_in[4];
  float* out     = (float*)d_out;
  float* meanSum = (float*)d_ws;

  const size_t needWs = WS_WQ_OFF + (size_t)GC * HID * sizeof(uint16_t);
  const bool pre = (ws_size >= needWs);

  gsq_zero<<<1, GC, 0, stream>>>(meanSum);
  if (pre) {
    uint32_t* wqb = (uint32_t*)((char*)d_ws + WS_WQ_OFF);
    gsq_prep<<<(GC * HID / 2) / 256, 256, 0, stream>>>(Wq, wqb);
    gsq_main<true><<<NROWS / ROWS_PER_WG, WAVES * 32, 0, stream>>>(
        x, gumbel, Wq, (const __bf16*)wqb, bq, cv, out, meanSum);
  } else {
    gsq_main<false><<<NROWS / ROWS_PER_WG, WAVES * 32, 0, stream>>>(
        x, gumbel, Wq, nullptr, bq, cv, out, meanSum);
  }
  gsq_finalize<<<1, GC, 0, stream>>>(meanSum, out, out_size - 1);
}
